// ArcFace_56710748176751
// MI455X (gfx1250) — matrix operand order repaired and checked by reading, not compiled, not run on hardware
//
#include <hip/hip_runtime.h>

// ---------------------------------------------------------------------------
// ArcFace loss for MI455X (gfx1250): bf16 hi/lo split WMMA GEMM + fused
// margin + streaming (online) softmax. Logits matrix [512 x 100000] is never
// materialized; weight (205 MB) is streamed from HBM exactly once.
// ---------------------------------------------------------------------------

typedef __bf16 v16bf __attribute__((ext_vector_type(16)));
typedef __bf16 v8bf  __attribute__((ext_vector_type(8)));
typedef float  v8f   __attribute__((ext_vector_type(8)));

constexpr int   kN = 512;
constexpr int   kD = 512;
constexpr int   kC = 100000;
constexpr float kScale = 64.0f;
constexpr float kEps   = 1e-7f;
constexpr float kCosM  = 0.8775825618903728f;   // cos(0.5)
constexpr float kSinM  = 0.4794255386042030f;   // sin(0.5)
constexpr float kTH    = -0.8775825618903728f;  // cos(pi - 0.5)
constexpr float kMM    = 0.2397127693021015f;   // sin(pi - 0.5) * 0.5

constexpr int kCols      = 64;                        // classes per workgroup tile
constexpr int kTiles     = (kC + kCols - 1) / kCols;  // 1563
constexpr int kLdsStride = kD + 8;                    // bf16 elems/class row (16B bank-shift pad)
constexpr int kSmemBytes = 2 * kCols * kLdsStride * (int)sizeof(__bf16); // 133120 B

// ---------------------------------------------------------------------------
// Kernel 1: L2-normalize input rows -> bf16 hi/lo in workspace.
// One wave per row, fully coalesced (lane + 32*i).
// ---------------------------------------------------------------------------
__global__ __launch_bounds__(256) void normalize_x_kernel(
    const float* __restrict__ x, __bf16* __restrict__ xhi, __bf16* __restrict__ xlo) {
  const int lane = threadIdx.x & 31;
  const int wave = threadIdx.x >> 5;
  const int row  = blockIdx.x * 8 + wave;
  const float* xr = x + (size_t)row * kD;

  float v[16];
  float ss = 0.0f;
#pragma unroll
  for (int i = 0; i < 16; ++i) {
    v[i] = xr[lane + 32 * i];
    ss += v[i] * v[i];
  }
#pragma unroll
  for (int m = 16; m >= 1; m >>= 1) ss += __shfl_xor(ss, m, 32);
  const float inv = 1.0f / fmaxf(sqrtf(ss), 1e-12f);
#pragma unroll
  for (int i = 0; i < 16; ++i) {
    const float xn = v[i] * inv;
    const __bf16 h = (__bf16)xn;
    xhi[(size_t)row * kD + lane + 32 * i] = h;
    xlo[(size_t)row * kD + lane + 32 * i] = (__bf16)(xn - (float)h);
  }
}

// ---------------------------------------------------------------------------
// Kernel 2: main fused tile kernel. One workgroup (256 thr = 8 waves) per
// 64-class tile. Normalizes weight tile into LDS (bf16 hi/lo), then each wave
// computes 16x16 WMMA accumulations for 4 row-blocks x 4 col-blocks with the
// bf16x2 split (3 WMMAs / K-step), applies the ArcFace margin at the target
// column, and emits per-(tile,row) online-softmax partials.
// ---------------------------------------------------------------------------
__global__ __launch_bounds__(256, 2) void arcface_tile_kernel(
    const float* __restrict__ weight, const int* __restrict__ target,
    const __bf16* __restrict__ xhi, const __bf16* __restrict__ xlo,
    float* __restrict__ pmax, float* __restrict__ psum, float* __restrict__ tlogit) {
  extern __shared__ __align__(16) char smem[];
  __bf16* whi = (__bf16*)smem;
  __bf16* wlo = whi + kCols * kLdsStride;

  const int tile    = blockIdx.x;
  const int colBase = tile * kCols;
  const int lane    = threadIdx.x & 31;
  const int wave    = threadIdx.x >> 5;

  // ---- Stage 1: stream weight tile (coalesced), normalize, stash bf16 hi/lo in LDS
  for (int rr = wave; rr < kCols; rr += 8) {
    const int c = colBase + rr;
    float v[16];
    float ss = 0.0f;
    if (c < kC) {
      const float* wr = weight + (size_t)c * kD;
#pragma unroll
      for (int i = 0; i < 16; ++i) {
        v[i] = wr[lane + 32 * i];
        ss += v[i] * v[i];
      }
    } else {
#pragma unroll
      for (int i = 0; i < 16; ++i) v[i] = 0.0f;
    }
#pragma unroll
    for (int m = 16; m >= 1; m >>= 1) ss += __shfl_xor(ss, m, 32);
    const float inv = (c < kC) ? (1.0f / fmaxf(sqrtf(ss), 1e-12f)) : 0.0f;
#pragma unroll
    for (int i = 0; i < 16; ++i) {
      const float xn = v[i] * inv;
      const __bf16 h = (__bf16)xn;
      whi[rr * kLdsStride + lane + 32 * i] = h;
      wlo[rr * kLdsStride + lane + 32 * i] = (__bf16)(xn - (float)h);
    }
  }
  __syncthreads();

  const int half = lane >> 4;  // which 16-lane half
  const int l16  = lane & 15;

  // ---- Stage 2: WMMA GEMM. Wave owns row-blocks {wave, wave+8, wave+16, wave+24},
  //      all 4 col-blocks of this tile (so per-row reduction stays in-wave).
  for (int jj = 0; jj < 4; ++jj) {
    const int rb = wave + 8 * jj;  // 0..31
    const int R  = rb * 16;

    v8f acc[4] = {};

    const __bf16* xh_row = xhi + (size_t)(R + l16) * kD;
    const __bf16* xl_row = xlo + (size_t)(R + l16) * kD;

    for (int k0 = 0; k0 < kD; k0 += 32) {
      // A fragment (16x32 bf16): lanes 0-15 -> K = k0+0..15,
      // lanes 16-31 -> K = k0+16..31 (contiguous, matches B).
      union { v16bf v; v8bf h[2]; } ah, al;
      ah.h[0] = *(const v8bf*)(xh_row + k0 + 16 * half);
      ah.h[1] = *(const v8bf*)(xh_row + k0 + 16 * half + 8);
      al.h[0] = *(const v8bf*)(xl_row + k0 + 16 * half);
      al.h[1] = *(const v8bf*)(xl_row + k0 + 16 * half + 8);
#pragma unroll
      for (int cb = 0; cb < 4; ++cb) {
        // B fragment (32x16 bf16): N = l16, K = k0 + 16*half + 0..15 (contig in LDS).
        const __bf16* bh_p = whi + (cb * 16 + l16) * kLdsStride + k0 + 16 * half;
        const __bf16* bl_p = wlo + (cb * 16 + l16) * kLdsStride + k0 + 16 * half;
        union { v16bf v; v8bf h[2]; } bh, bl;
        bh.h[0] = *(const v8bf*)(bh_p);
        bh.h[1] = *(const v8bf*)(bh_p + 8);
        bl.h[0] = *(const v8bf*)(bl_p);
        bl.h[1] = *(const v8bf*)(bl_p + 8);
        // bf16x2 split: xh*wh + xl*wh + xh*wl  (fp32 accumulate)
        acc[cb] = __builtin_amdgcn_wmma_f32_16x16x32_bf16(
            false, ah.v, false, bh.v, (short)0, acc[cb], false, false);
        acc[cb] = __builtin_amdgcn_wmma_f32_16x16x32_bf16(
            false, al.v, false, bh.v, (short)0, acc[cb], false, false);
        acc[cb] = __builtin_amdgcn_wmma_f32_16x16x32_bf16(
            false, ah.v, false, bl.v, (short)0, acc[cb], false, false);
      }
    }

    // ---- Stage 3: margin + per-row online softmax partials.
    // C/D layout: element e of lane -> row M = 8*half + e, col N = l16.
#pragma unroll
    for (int e = 0; e < 8; ++e) {
      const int n   = R + 8 * half + e;
      const int tgt = target[n];
      float lm = -3.0e38f;
      float ls = 0.0f;
#pragma unroll
      for (int cb = 0; cb < 4; ++cb) {
        const int c = colBase + cb * 16 + l16;
        if (c < kC) {
          float cosv = acc[cb][e];
          cosv = fminf(fmaxf(cosv, -1.0f + kEps), 1.0f - kEps);
          float logit = cosv * kScale;
          if (c == tgt) {
            float s2 = 1.0f - cosv * cosv;
            s2 = fminf(fmaxf(s2, -1.0f + kEps), 1.0f - kEps);
            const float sine = sqrtf(s2);
            float phi = cosv * kCosM - sine * kSinM;
            if (!(cosv > kTH)) phi = cosv - kMM;
            logit = phi * kScale;
            tlogit[n] = logit;  // exactly one writer per row across the grid
          }
          const float M = fmaxf(lm, logit);
          ls = ls * __expf(lm - M) + __expf(logit - M);
          lm = M;
        }
      }
      // reduce (max, sumexp) across the 16 lanes of this half-wave
#pragma unroll
      for (int msk = 1; msk <= 8; msk <<= 1) {
        const float om = __shfl_xor(lm, msk, 32);
        const float os = __shfl_xor(ls, msk, 32);
        const float M  = fmaxf(lm, om);
        ls = ls * __expf(lm - M) + os * __expf(om - M);
        lm = M;
      }
      if (l16 == 0) {
        pmax[(size_t)tile * kN + n] = lm;
        psum[(size_t)tile * kN + n] = ls;
      }
    }
  }
}

// ---------------------------------------------------------------------------
// Kernel 3: per-row log-sum-exp combine over tile partials; row loss.
// ---------------------------------------------------------------------------
__global__ __launch_bounds__(256) void reduce_rows_kernel(
    const float* __restrict__ pmax, const float* __restrict__ psum,
    const float* __restrict__ tlogit, float* __restrict__ rowloss) {
  const int n   = blockIdx.x;
  const int tid = threadIdx.x;
  float lm = -3.0e38f, ls = 0.0f;
  for (int t = tid; t < kTiles; t += 256) {
    const float m = pmax[(size_t)t * kN + n];
    const float s = psum[(size_t)t * kN + n];
    const float M = fmaxf(lm, m);
    ls = ls * __expf(lm - M) + s * __expf(m - M);
    lm = M;
  }
#pragma unroll
  for (int msk = 1; msk < 32; msk <<= 1) {
    const float om = __shfl_xor(lm, msk, 32);
    const float os = __shfl_xor(ls, msk, 32);
    const float M  = fmaxf(lm, om);
    ls = ls * __expf(lm - M) + os * __expf(om - M);
    lm = M;
  }
  __shared__ float smx[8];
  __shared__ float ssm[8];
  if ((tid & 31) == 0) { smx[tid >> 5] = lm; ssm[tid >> 5] = ls; }
  __syncthreads();
  if (tid == 0) {
    lm = smx[0]; ls = ssm[0];
    for (int i = 1; i < 8; ++i) {
      const float m = smx[i], s = ssm[i];
      const float M = fmaxf(lm, m);
      ls = ls * __expf(lm - M) + s * __expf(m - M);
      lm = M;
    }
    rowloss[n] = (lm + logf(ls)) - tlogit[n];
  }
}

// ---------------------------------------------------------------------------
// Kernel 4: mean over 512 row losses -> scalar output.
// ---------------------------------------------------------------------------
__global__ __launch_bounds__(256) void final_mean_kernel(
    const float* __restrict__ rowloss, float* __restrict__ out) {
  __shared__ float s[256];
  const int tid = threadIdx.x;
  s[tid] = rowloss[tid] + rowloss[tid + 256];
  __syncthreads();
  for (int st = 128; st > 0; st >>= 1) {
    if (tid < st) s[tid] += s[tid + st];
    __syncthreads();
  }
  if (tid == 0) out[0] = s[0] / (float)kN;
}

// ---------------------------------------------------------------------------
// Launcher. Inputs (setup_inputs order): input f32 [512,512], target int [512],
// weight f32 [100000,512]. Output: scalar f32 loss.
// ---------------------------------------------------------------------------
extern "C" void kernel_launch(void* const* d_in, const int* in_sizes, int n_in,
                              void* d_out, int out_size, void* d_ws, size_t ws_size,
                              hipStream_t stream) {
  (void)in_sizes; (void)n_in; (void)out_size; (void)ws_size;
  const float* input  = (const float*)d_in[0];
  const int*   target = (const int*)d_in[1];
  const float* weight = (const float*)d_in[2];
  float* out = (float*)d_out;

  // Workspace layout (bytes):
  //   [0, 512K)          x_hi  bf16 [512][512]
  //   [512K, 1M)         x_lo  bf16 [512][512]
  //   [1M, 1M+4K)        tlogit f32 [512]
  //   [1M+4K, 1M+8K)     rowloss f32 [512]
  //   [1M+8K, ...)       pmax f32 [kTiles][512], then psum f32 [kTiles][512]
  char* ws = (char*)d_ws;
  __bf16* xhi  = (__bf16*)(ws);
  __bf16* xlo  = (__bf16*)(ws + (size_t)512 * 1024);
  float*  tlog = (float*)(ws + (size_t)1024 * 1024);
  float*  rls  = (float*)(ws + (size_t)1024 * 1024 + 4096);
  float*  pmax = (float*)(ws + (size_t)1024 * 1024 + 8192);
  float*  psum = pmax + (size_t)kTiles * kN;

  // 130 KB dynamic LDS per workgroup (gfx1250 WGP has 320 KB).
  hipFuncSetAttribute(reinterpret_cast<const void*>(arcface_tile_kernel),
                      hipFuncAttributeMaxDynamicSharedMemorySize, kSmemBytes);

  normalize_x_kernel<<<kN / 8, 256, 0, stream>>>(input, xhi, xlo);
  arcface_tile_kernel<<<kTiles, 256, kSmemBytes, stream>>>(
      weight, target, xhi, xlo, pmax, psum, tlog);
  reduce_rows_kernel<<<kN, 256, 0, stream>>>(pmax, psum, tlog, rls);
  final_mean_kernel<<<1, 256, 0, stream>>>(rls, out);
}
